// SceneGlue_66864050864862
// MI455X (gfx1250) — matrix operand order repaired and checked by reading, not compiled, not run on hardware
//
#include <hip/hip_runtime.h>
#include <hip/hip_bf16.h>
#include <cmath>

// ---------------- problem constants ----------------
#define BB      4
#define NPTS    1024
#define CC      256
#define NGT     2
#define RPB     1026          // rows per (batch,side) = NPTS + NGT
#define VB      8             // 2*B virtual batches (x half: 0..3, y half: 4..7)
#define TOTROWS (VB*RPB)      // 8208
#define HEADS   4
#define DH      64
#define HID     512
#define NP1     1025          // N+1 (sinkhorn size)
#define SINK_IT 40

typedef __attribute__((ext_vector_type(16))) _Float16 v16h;
typedef __attribute__((ext_vector_type(8)))  _Float16 v8h;
typedef __attribute__((ext_vector_type(8)))  float    v8f;

__device__ __forceinline__ v16h cat8(v8h lo, v8h hi) {
    return __builtin_shufflevector(lo, hi, 0,1,2,3,4,5,6,7,8,9,10,11,12,13,14,15);
}
__device__ __forceinline__ v8f wmma32(v16h a, v16h b, v8f c) {
    return __builtin_amdgcn_wmma_f32_16x16x32_f16(false, a, false, b, (short)0, c, false, false);
}
__device__ __forceinline__ float gelu_f(float x) {
    return 0.5f * x * (1.f + tanhf(0.7978845608028654f * (x + 0.044715f * x * x * x)));
}

// CDNA5 async global->LDS copy (ASYNCcnt-tracked, GVS addressing mode).
// vdst = LDS byte address (low 32 bits of flat shared pointer, per aperture
// truncation rule), vaddr = per-lane byte offset, saddr = uniform 64-bit base.
__device__ __forceinline__ void async_b128(void* lds, unsigned goff,
                                           unsigned long long sbase) {
    unsigned l = (unsigned)(unsigned long long)lds;
    asm volatile("global_load_async_to_lds_b128 %0, %1, %2"
                 :: "v"(l), "v"(goff), "s"(sbase) : "memory");
}
__device__ __forceinline__ void wait_async0() {
    asm volatile("s_wait_asynccnt 0" ::: "memory");
}

// =====================================================================
// kenc MLP (3->32->64->128->256) + desc^T add, written directly into xy
// =====================================================================
template<int DIN, int DOUT, bool RELU>
__device__ __forceinline__ void kenc_layer(const float* __restrict__ w,
                                           const float* __restrict__ bias,
                                           const float* src, float* dst, int lane) {
    const int NPL = DOUT / 32;
    float acc[NPL];
#pragma unroll
    for (int t = 0; t < NPL; ++t) acc[t] = bias[lane + 32 * t];
    for (int i = 0; i < DIN; ++i) {
        float xv = src[i];
#pragma unroll
        for (int t = 0; t < NPL; ++t) acc[t] += xv * w[i * DOUT + lane + 32 * t];
    }
#pragma unroll
    for (int t = 0; t < NPL; ++t) dst[lane + 32 * t] = RELU ? fmaxf(acc[t], 0.f) : acc[t];
}

__global__ __launch_bounds__(128) void kenc_kernel(
    const float* __restrict__ desc0, const float* __restrict__ desc1,
    const float* __restrict__ kpts0, const float* __restrict__ kpts1,
    const float* __restrict__ sc0,   const float* __restrict__ sc1,
    const float* kb0, const float* kw0, const float* kb1, const float* kw1,
    const float* kb2, const float* kw2, const float* kb3, const float* kw3,
    float* __restrict__ xy)
{
    __shared__ float buf[4][2][CC];
    int tid  = threadIdx.x;
    int wid  = tid >> 5;
    int lane = tid & 31;
    int p    = blockIdx.x * 4 + wid;          // 0..8191
    int vb   = p >> 10;
    int i    = p & 1023;
    int side = vb >> 2;
    int b    = vb & 3;

    const float* kp = side ? kpts1 : kpts0;
    const float* sc = side ? sc1   : sc0;
    if (lane == 0) {
        buf[wid][0][0] = (kp[(b * NPTS + i) * 2 + 0] - 320.f) / 448.f;
        buf[wid][0][1] = (kp[(b * NPTS + i) * 2 + 1] - 240.f) / 448.f;
        buf[wid][0][2] = sc[b * NPTS + i];
    }
    __syncthreads();
    kenc_layer<3,   32, true >(kw0, kb0, buf[wid][0], buf[wid][1], lane); __syncthreads();
    kenc_layer<32,  64, true >(kw1, kb1, buf[wid][1], buf[wid][0], lane); __syncthreads();
    kenc_layer<64, 128, true >(kw2, kb2, buf[wid][0], buf[wid][1], lane); __syncthreads();
    kenc_layer<128,256, false>(kw3, kb3, buf[wid][1], buf[wid][0], lane); __syncthreads();

    const float* dsc = side ? desc1 : desc0;  // (B, C, N)
    float* orow = xy + (size_t)(vb * RPB + i) * CC;
#pragma unroll
    for (int t = 0; t < 8; ++t) {
        int c = lane + 32 * t;
        orow[c] = dsc[(size_t)(b * CC + c) * NPTS + i] + buf[wid][0][c];
    }
}

__global__ __launch_bounds__(256) void set_gt_kernel(const float* __restrict__ gt,
                                                     float* __restrict__ xy)
{
    int idx = blockIdx.x * 256 + threadIdx.x;           // VB*NGT*CC = 4096
    if (idx >= VB * NGT * CC) return;
    int c  = idx & 255;
    int g  = (idx >> 8) & 1;
    int vb = idx >> 9;
    xy[(size_t)(vb * RPB + NPTS + g) * CC + c] = gt[g * CC + c];
}

// =====================================================================
// LayerNorm over C=256, one wave per row
// =====================================================================
__global__ __launch_bounds__(256) void ln_kernel(const float* __restrict__ in,
                                                 float* __restrict__ out,
                                                 const float* __restrict__ gg,
                                                 const float* __restrict__ bb,
                                                 int rows)
{
    int wid  = threadIdx.x >> 5;
    int lane = threadIdx.x & 31;
    int row  = blockIdx.x * 8 + wid;
    if (row >= rows) return;
    const float* r = in + (size_t)row * CC;
    float x[8], s = 0.f;
#pragma unroll
    for (int k = 0; k < 8; ++k) { x[k] = r[lane + 32 * k]; s += x[k]; }
#pragma unroll
    for (int m = 16; m >= 1; m >>= 1) s += __shfl_xor(s, m, 32);
    float mu = s * (1.f / 256.f);
    float v = 0.f;
#pragma unroll
    for (int k = 0; k < 8; ++k) { float d = x[k] - mu; v += d * d; }
#pragma unroll
    for (int m = 16; m >= 1; m >>= 1) v += __shfl_xor(v, m, 32);
    float rs = rsqrtf(v * (1.f / 256.f) + 1e-5f);
    float* o = out + (size_t)row * CC;
#pragma unroll
    for (int k = 0; k < 8; ++k) {
        int c = lane + 32 * k;
        o[c] = (x[k] - mu) * rs * gg[c] + bb[c];
    }
}

// =====================================================================
// WMMA GEMM with CDNA5 async global->LDS staging + LDS ping-pong.
// out = act(A @ W + bias) [+ residual]; A: RxK f32, W: KxN f32, tile 64x64.
// Both tiles stay row-major in LDS (f32); f32->f16 happens at fragment
// assembly, so the copy itself never touches VGPRs (ASYNCcnt path).
// =====================================================================
__global__ __launch_bounds__(128) void gemm_kernel(
    const float* __restrict__ A, const float* __restrict__ W,
    const float* __restrict__ bias, const float* __restrict__ res,
    float* __restrict__ out, int R, int K, int N, int act)
{
    __shared__ float sAf[2][64 * 32];
    __shared__ float sWf[2][32 * 64];
    int tid  = threadIdx.x;
    int wid  = tid >> 5;
    int lane = tid & 31;
    int m    = lane & 15;
    int g2   = lane >> 4;
    int r0   = blockIdx.x * 64;
    int n0   = blockIdx.y * 64;
    unsigned long long baseA = (unsigned long long)A;
    unsigned long long baseW = (unsigned long long)W;

    auto issue = [&](int bufI, int k0) {
        // A tile: 64 rows x 32 f32 -> 512 16-byte chunks
        for (int c = tid; c < 512; c += 128) {
            int r = c >> 3, q = c & 7;
            int gr = r0 + r;
            if (gr < R) {
                unsigned go = (unsigned)(((size_t)gr * K + k0 + q * 4) * 4);
                async_b128(&sAf[bufI][r * 32 + q * 4], go, baseA);
            } else {
                float4 z = {0.f, 0.f, 0.f, 0.f};
                *(float4*)&sAf[bufI][r * 32 + q * 4] = z;
            }
        }
        // W tile: 32 rows x 64 f32 -> 512 16-byte chunks
        for (int c = tid; c < 512; c += 128) {
            int kk = c >> 4, q = c & 15;
            unsigned go = (unsigned)(((size_t)(k0 + kk) * N + n0 + q * 4) * 4);
            async_b128(&sWf[bufI][kk * 64 + q * 4], go, baseW);
        }
    };

    v8f acc[4];
#pragma unroll
    for (int t = 0; t < 4; ++t)
#pragma unroll
        for (int e = 0; e < 8; ++e) acc[t][e] = 0.f;

    const int KT = K >> 5;
    issue(0, 0);
    int cur = 0;
    for (int kt = 0; kt < KT; ++kt) {
        wait_async0();            // our async copies are invisible to the compiler
        __syncthreads();          // tile[cur] ready everywhere; prev compute done
        if (kt + 1 < KT) issue(cur ^ 1, (kt + 1) * 32);

        const float* pa = &sAf[cur][(wid * 16 + m) * 32];
        v16h af;
#pragma unroll
        for (int e = 0; e < 8; ++e) {
            af[e]     = (_Float16)pa[g2 * 8 + e];
            af[8 + e] = (_Float16)pa[16 + g2 * 8 + e];
        }
#pragma unroll
        for (int t = 0; t < 4; ++t) {
            const float* pb = &sWf[cur][t * 16 + m];
            v16h bf;
#pragma unroll
            for (int e = 0; e < 16; ++e) bf[e] = (_Float16)pb[((e >> 3) * 16 + g2 * 8 + (e & 7)) * 64];
            acc[t] = wmma32(af, bf, acc[t]);
        }
        cur ^= 1;
    }
#pragma unroll
    for (int t = 0; t < 4; ++t) {
        int col = n0 + t * 16 + m;
        float bv = bias[col];
#pragma unroll
        for (int v = 0; v < 8; ++v) {
            int grow = r0 + wid * 16 + v + 8 * g2;
            if (grow < R) {
                float val = acc[t][v] + bv;
                if (act == 1) val = fmaxf(val, 0.f);
                else if (act == 2) val = gelu_f(val);
                if (res) val += res[(size_t)grow * N + col];
                out[(size_t)grow * N + col] = val;
            }
        }
    }
}

// =====================================================================
// Flash attention with WMMA (16 rows per wave, 32-key chunks)
// grid (ceil(1026/64), HEADS, VB); kvXor flips half for cross-attn
// =====================================================================
__global__ __launch_bounds__(128) void attn_kernel(
    const float* __restrict__ qb, const float* __restrict__ kb,
    const float* __restrict__ vbuf, float* __restrict__ ob, int kvXor)
{
    __shared__ _Float16 sKT[64 * 32];        // [dh][key] (transposed on store)
    __shared__ _Float16 sV[32 * 64];         // [key][dh]
    __shared__ _Float16 sP[4][16 * 32];      // per-wave P staging
    int tid  = threadIdx.x;
    int wid  = tid >> 5;
    int lane = tid & 31;
    int m    = lane & 15;
    int g2   = lane >> 4;
    int vb   = blockIdx.z;
    int h    = blockIdx.y;
    int qB   = vb * RPB;
    int kvB  = (vb ^ kvXor) * RPB;
    int tile0 = blockIdx.x * 64 + wid * 16;

    // Q fragments (K-dim 0..31 and 32..63), prescaled by 1/sqrt(dh)
    int qrow = tile0 + m;
    bool rv = qrow < RPB;
    v16h qf[2];
#pragma unroll
    for (int kk = 0; kk < 2; ++kk)
#pragma unroll
        for (int e = 0; e < 16; ++e) {
            int kd = kk * 32 + (e >> 3) * 16 + g2 * 8 + (e & 7);
            float val = rv ? qb[(size_t)(qB + qrow) * CC + h * DH + kd] * 0.125f : 0.f;
            qf[kk][e] = (_Float16)val;
        }

    float mrow[8], lsum[8];
#pragma unroll
    for (int v = 0; v < 8; ++v) { mrow[v] = -1e30f; lsum[v] = 0.f; }
    v8f acc[4];
#pragma unroll
    for (int t = 0; t < 4; ++t)
#pragma unroll
        for (int e = 0; e < 8; ++e) acc[t][e] = 0.f;

    for (int key0 = 0; key0 < RPB; key0 += 32) {
        // prefetch next chunk of K/V rows (global_prefetch_b8 path)
        if (key0 + 32 < RPB) {
            int pk = kvB + key0 + 32 + (tid >> 2);
            __builtin_prefetch(&kb[(size_t)pk * CC + h * DH], 0, 0);
            __builtin_prefetch(&vbuf[(size_t)pk * CC + h * DH], 0, 0);
        }
        __syncthreads();
        for (int idx = tid; idx < 2048; idx += 128) {
            int dh = idx & 63, ky = idx >> 6;
            int gk = key0 + ky;
            float kvv = (gk < RPB) ? kb[(size_t)(kvB + gk) * CC + h * DH + dh] : 0.f;
            float vvv = (gk < RPB) ? vbuf[(size_t)(kvB + gk) * CC + h * DH + dh] : 0.f;
            sKT[dh * 32 + ky] = (_Float16)kvv;
            sV[ky * 64 + dh]  = (_Float16)vvv;
        }
        __syncthreads();

        v8f s[2];
#pragma unroll
        for (int t = 0; t < 2; ++t) {
#pragma unroll
            for (int e = 0; e < 8; ++e) s[t][e] = 0.f;
#pragma unroll
            for (int kk = 0; kk < 2; ++kk) {
                v16h bk;
                for (int e = 0; e < 16; ++e)
                    bk[e] = sKT[(kk * 32 + (e >> 3) * 16 + g2 * 8 + (e & 7)) * 32 + t * 16 + m];
                s[t] = wmma32(qf[kk], bk, s[t]);
            }
            int keyc = key0 + t * 16 + m;
            if (keyc >= RPB) {
#pragma unroll
                for (int v = 0; v < 8; ++v) s[t][v] = -1e30f;
            }
        }
        // online softmax (rows striped: row = v + 8*g2, cols across 16-lane group)
        float rm[8];
#pragma unroll
        for (int v = 0; v < 8; ++v) rm[v] = fmaxf(s[0][v], s[1][v]);
#pragma unroll
        for (int mk = 1; mk < 16; mk <<= 1)
#pragma unroll
            for (int v = 0; v < 8; ++v) rm[v] = fmaxf(rm[v], __shfl_xor(rm[v], mk, 32));
        float corr[8], rs[8];
#pragma unroll
        for (int v = 0; v < 8; ++v) {
            float mn = fmaxf(mrow[v], rm[v]);
            corr[v]  = expf(mrow[v] - mn);
            mrow[v]  = mn;
            float p0 = expf(s[0][v] - mn);
            float p1 = expf(s[1][v] - mn);
            s[0][v] = p0; s[1][v] = p1;
            rs[v] = p0 + p1;
        }
#pragma unroll
        for (int mk = 1; mk < 16; mk <<= 1)
#pragma unroll
            for (int v = 0; v < 8; ++v) rs[v] += __shfl_xor(rs[v], mk, 32);
#pragma unroll
        for (int v = 0; v < 8; ++v) {
            lsum[v] = lsum[v] * corr[v] + rs[v];
#pragma unroll
            for (int t = 0; t < 4; ++t) acc[t][v] *= corr[v];
        }
        // stage P as f16 A-fragment source
#pragma unroll
        for (int t = 0; t < 2; ++t)
#pragma unroll
            for (int v = 0; v < 8; ++v)
                sP[wid][(v + 8 * g2) * 32 + t * 16 + m] = (_Float16)s[t][v];
        __syncthreads();
        v8h plo = *(const v8h*)&sP[wid][m * 32 + g2 * 8];
        v8h phi = *(const v8h*)&sP[wid][m * 32 + 16 + g2 * 8];
        v16h pf = cat8(plo, phi);
#pragma unroll
        for (int tt = 0; tt < 4; ++tt) {
            v16h bv;
            for (int e = 0; e < 16; ++e)
                bv[e] = sV[((e >> 3) * 16 + g2 * 8 + (e & 7)) * 64 + tt * 16 + m];
            acc[tt] = wmma32(pf, bv, acc[tt]);
        }
    }
#pragma unroll
    for (int tt = 0; tt < 4; ++tt)
#pragma unroll
        for (int v = 0; v < 8; ++v) {
            int row = tile0 + v + 8 * g2;
            if (row < RPB)
                ob[(size_t)(qB + row) * CC + h * DH + tt * 16 + m] = acc[tt][v] / lsum[v];
        }
}

// =====================================================================
// scores[b,i,j] = dot(md0[b,i,:], md1[b,j,:]) / 16   (A @ B^T via WMMA;
// B tile is transposed on store so it keeps the synchronous path)
// =====================================================================
__global__ __launch_bounds__(128) void scores_kernel(const float* __restrict__ md,
                                                     float* __restrict__ Z)
{
    __shared__ _Float16 sA[64 * 32];
    __shared__ _Float16 sBT[32 * 64];
    int tid  = threadIdx.x;
    int wid  = tid >> 5;
    int lane = tid & 31;
    int m    = lane & 15;
    int g2   = lane >> 4;
    int b    = blockIdx.z;
    int i0   = blockIdx.x * 64;
    int j0   = blockIdx.y * 64;
    size_t base0 = (size_t)(b * RPB) * CC;
    size_t base1 = (size_t)((b + 4) * RPB) * CC;

    v8f acc[4];
#pragma unroll
    for (int t = 0; t < 4; ++t)
#pragma unroll
        for (int e = 0; e < 8; ++e) acc[t][e] = 0.f;

    for (int c0 = 0; c0 < CC; c0 += 32) {
        __syncthreads();
        for (int idx = tid; idx < 2048; idx += 128) {
            int r = idx >> 5, cc = idx & 31;
            sA[r * 32 + cc] = (_Float16)md[base0 + (size_t)(i0 + r) * CC + c0 + cc];
        }
        for (int idx = tid; idx < 2048; idx += 128) {
            int cc = idx & 31, j = idx >> 5;
            sBT[cc * 64 + j] = (_Float16)md[base1 + (size_t)(j0 + j) * CC + c0 + cc];
        }
        __syncthreads();
        v8h alo = *(const v8h*)&sA[(wid * 16 + m) * 32 + g2 * 8];
        v8h ahi = *(const v8h*)&sA[(wid * 16 + m) * 32 + 16 + g2 * 8];
        v16h af = cat8(alo, ahi);
#pragma unroll
        for (int t = 0; t < 4; ++t) {
            v16h bs;
            for (int e = 0; e < 16; ++e)
                bs[e] = sBT[((e >> 3) * 16 + g2 * 8 + (e & 7)) * 64 + t * 16 + m];
            acc[t] = wmma32(af, bs, acc[t]);
        }
    }
    size_t zb = (size_t)b * NP1 * NP1;
#pragma unroll
    for (int t = 0; t < 4; ++t)
#pragma unroll
        for (int v = 0; v < 8; ++v) {
            int gi = i0 + wid * 16 + v + 8 * g2;
            int gj = j0 + t * 16 + m;
            Z[zb + (size_t)gi * NP1 + gj] = acc[t][v] * 0.0625f;
        }
}

// =====================================================================
// Sinkhorn
// =====================================================================
__global__ __launch_bounds__(256) void bins_kernel(float* __restrict__ Z,
                                                   const float* __restrict__ alpha_p)
{
    int idx = blockIdx.x * 256 + threadIdx.x;
    if (idx >= BB * 2050) return;
    float a = alpha_p[0];
    int b = idx / 2050, r = idx % 2050;
    size_t zb = (size_t)b * NP1 * NP1;
    if (r < NP1) Z[zb + (size_t)r * NP1 + NPTS] = a;
    else         Z[zb + (size_t)NPTS * NP1 + (r - NP1)] = a;
}

__global__ __launch_bounds__(256) void uv_init_kernel(float* __restrict__ u,
                                                      float* __restrict__ v)
{
    int idx = blockIdx.x * 256 + threadIdx.x;
    if (idx < BB * NP1) { u[idx] = 0.f; v[idx] = 0.f; }
}

__global__ __launch_bounds__(256) void sink_u_kernel(const float* __restrict__ Z,
                                                     const float* __restrict__ v,
                                                     float* __restrict__ u,
                                                     float norm, float logn)
{
    __shared__ float rm[256], rs[256];
    int b = blockIdx.x / NP1, row = blockIdx.x % NP1;
    const float* zr = Z + ((size_t)b * NP1 + row) * NP1;
    const float* vr = v + b * NP1;
    float mloc = -1e30f, sloc = 0.f;
    for (int j = threadIdx.x; j < NP1; j += 256) {
        float z = zr[j] + vr[j];
        float nm = fmaxf(mloc, z);
        sloc = sloc * expf(mloc - nm) + expf(z - nm);
        mloc = nm;
    }
    rm[threadIdx.x] = mloc; rs[threadIdx.x] = sloc;
    __syncthreads();
    for (int off = 128; off >= 1; off >>= 1) {
        if (threadIdx.x < off) {
            float m2 = rm[threadIdx.x + off], s2 = rs[threadIdx.x + off];
            float m1 = rm[threadIdx.x],       s1 = rs[threadIdx.x];
            float nm = fmaxf(m1, m2);
            rs[threadIdx.x] = s1 * expf(m1 - nm) + s2 * expf(m2 - nm);
            rm[threadIdx.x] = nm;
        }
        __syncthreads();
    }
    if (threadIdx.x == 0) {
        float mu = (row < NPTS) ? norm : logn;
        u[b * NP1 + row] = mu - (rm[0] + logf(rs[0]));
    }
}

__global__ __launch_bounds__(128) void sink_v_kernel(const float* __restrict__ Z,
                                                     const float* __restrict__ u,
                                                     float* __restrict__ v,
                                                     float norm, float logn)
{
    int b = blockIdx.x;
    int col = blockIdx.y * 128 + threadIdx.x;
    if (col >= NP1) return;
    const float* ur = u + b * NP1;
    size_t zb = (size_t)b * NP1 * NP1;
    float mloc = -1e30f, sloc = 0.f;
    for (int i = 0; i < NP1; ++i) {
        float z = Z[zb + (size_t)i * NP1 + col] + ur[i];
        float nm = fmaxf(mloc, z);
        sloc = sloc * expf(mloc - nm) + expf(z - nm);
        mloc = nm;
    }
    float nu = (col < NPTS) ? norm : logn;
    v[b * NP1 + col] = nu - (mloc + logf(sloc));
}

__global__ __launch_bounds__(256) void zfinal_kernel(float* __restrict__ Z,
                                                     const float* __restrict__ u,
                                                     const float* __restrict__ v,
                                                     float* __restrict__ expZ,
                                                     float norm)
{
    size_t idx = (size_t)blockIdx.x * 256 + threadIdx.x;
    if (idx >= (size_t)BB * NP1 * NP1) return;
    int b = (int)(idx / (NP1 * NP1));
    int rem = (int)(idx % (NP1 * NP1));
    int i = rem / NP1, j = rem % NP1;
    float val = Z[idx] + u[b * NP1 + i] + v[b * NP1 + j] - norm;
    Z[idx] = val;
    expZ[idx] = expf(val);
}

// =====================================================================
// matching
// =====================================================================
__global__ __launch_bounds__(256) void rowmax_kernel(const float* __restrict__ Z,
                                                     float* __restrict__ mx,
                                                     int* __restrict__ ix)
{
    __shared__ float rv[256];
    __shared__ int   ri[256];
    int b = blockIdx.x / NPTS, i = blockIdx.x % NPTS;
    const float* zr = Z + ((size_t)b * NP1 + i) * NP1;
    float mv = -1e30f; int mi = 0;
    for (int j = threadIdx.x; j < NPTS; j += 256)
        if (zr[j] > mv) { mv = zr[j]; mi = j; }
    rv[threadIdx.x] = mv; ri[threadIdx.x] = mi;
    __syncthreads();
    for (int off = 128; off >= 1; off >>= 1) {
        if (threadIdx.x < off) {
            if (rv[threadIdx.x + off] > rv[threadIdx.x]) {
                rv[threadIdx.x] = rv[threadIdx.x + off];
                ri[threadIdx.x] = ri[threadIdx.x + off];
            }
        }
        __syncthreads();
    }
    if (threadIdx.x == 0) { mx[b * NPTS + i] = rv[0]; ix[b * NPTS + i] = ri[0]; }
}

__global__ __launch_bounds__(128) void colmax_kernel(const float* __restrict__ Z,
                                                     float* __restrict__ mx,
                                                     int* __restrict__ ix)
{
    int b = blockIdx.x;
    int col = blockIdx.y * 128 + threadIdx.x;
    if (col >= NPTS) return;
    size_t zb = (size_t)b * NP1 * NP1;
    float mv = -1e30f; int mi = 0;
    for (int i = 0; i < NPTS; ++i) {
        float z = Z[zb + (size_t)i * NP1 + col];
        if (z > mv) { mv = z; mi = i; }
    }
    mx[b * NPTS + col] = mv; ix[b * NPTS + col] = mi;
}

__global__ __launch_bounds__(256) void match0_kernel(const float* __restrict__ max0,
                                                     const int* __restrict__ idx0,
                                                     const int* __restrict__ idx1,
                                                     float* __restrict__ out_ms0,
                                                     float* __restrict__ out_i0,
                                                     int* __restrict__ valid0)
{
    int idx = blockIdx.x * 256 + threadIdx.x;
    if (idx >= BB * NPTS) return;
    int b = idx / NPTS, i = idx % NPTS;
    int j = idx0[idx];
    bool mutual = (idx1[b * NPTS + j] == i);
    float ms = mutual ? expf(max0[idx]) : 0.f;
    bool val = mutual && (ms > 0.2f);
    out_ms0[idx] = ms;
    out_i0[idx]  = val ? (float)j : -1.f;
    valid0[idx]  = val ? 1 : 0;
}

__global__ __launch_bounds__(256) void match1_kernel(const int* __restrict__ idx0,
                                                     const int* __restrict__ idx1,
                                                     const float* __restrict__ ms0,
                                                     const int* __restrict__ valid0,
                                                     float* __restrict__ out_ms1,
                                                     float* __restrict__ out_i1)
{
    int idx = blockIdx.x * 256 + threadIdx.x;
    if (idx >= BB * NPTS) return;
    int b = idx / NPTS, j = idx % NPTS;
    int i = idx1[idx];
    bool mutual = (idx0[b * NPTS + i] == j);
    float ms = mutual ? ms0[b * NPTS + i] : 0.f;
    bool val = mutual && (valid0[b * NPTS + i] != 0);
    out_ms1[idx] = ms;
    out_i1[idx]  = val ? (float)i : -1.f;
}

// =====================================================================
// grouping attention (tiny: 2 queries x 1024 keys per (vb, head))
// =====================================================================
__global__ __launch_bounds__(256) void gattn_kernel(const float* __restrict__ qb,
                                                    const float* __restrict__ kb,
                                                    const float* __restrict__ vbuf,
                                                    float* __restrict__ gatt)
{
    __shared__ float sS[2 * NPTS];
    __shared__ float sM[2], sSum[2];
    int vb = blockIdx.x, h = blockIdx.y;
    int qbase = vb * RPB + NPTS;
    int kbase = vb * RPB;
    for (int idx = threadIdx.x; idx < 2 * NPTS; idx += 256) {
        int q = idx >> 10, key = idx & 1023;
        float acc = 0.f;
        for (int d = 0; d < DH; ++d)
            acc += qb[(size_t)(qbase + q) * CC + h * DH + d] *
                   kb[(size_t)(kbase + key) * CC + h * DH + d];
        sS[idx] = acc * 0.125f;
    }
    __syncthreads();
    if (threadIdx.x < 2) {
        int q = threadIdx.x;
        float mv = -1e30f;
        for (int k = 0; k < NPTS; ++k) mv = fmaxf(mv, sS[q * NPTS + k]);
        float s = 0.f;
        for (int k = 0; k < NPTS; ++k) s += expf(sS[q * NPTS + k] - mv);
        sM[q] = mv; sSum[q] = s;
    }
    __syncthreads();
    if (threadIdx.x < 128) {
        int q = threadIdx.x >> 6, d = threadIdx.x & 63;
        float acc = 0.f;
        for (int k = 0; k < NPTS; ++k)
            acc += expf(sS[q * NPTS + k] - sM[q]) *
                   vbuf[(size_t)(kbase + k) * CC + h * DH + d];
        gatt[(size_t)(vb * 2 + q) * CC + h * DH + d] = acc / sSum[q];
    }
}

__global__ __launch_bounds__(256) void gout_kernel(const float* __restrict__ gatt,
                                                   const float* __restrict__ wo,
                                                   const float* __restrict__ bo,
                                                   const float* __restrict__ xy,
                                                   float* __restrict__ d_out,
                                                   size_t offG0, size_t offG1)
{
    int vb = blockIdx.x, q = blockIdx.y, c = threadIdx.x;
    float acc = bo[c];
    const float* g = gatt + (size_t)(vb * 2 + q) * CC;
    for (int d = 0; d < CC; ++d) acc += g[d] * wo[d * CC + c];
    float res = xy[(size_t)(vb * RPB + NPTS + q) * CC + c] + acc;
    if (vb < 4) d_out[offG0 + (size_t)(vb * 2 + q) * CC + c] = res;
    else        d_out[offG1 + (size_t)((vb - 4) * 2 + q) * CC + c] = res;
}

// =====================================================================
// host launch
// =====================================================================
extern "C" void kernel_launch(void* const* d_in, const int* in_sizes, int n_in,
                              void* d_out, int out_size, void* d_ws, size_t ws_size,
                              hipStream_t stream)
{
    (void)in_sizes; (void)n_in; (void)out_size; (void)ws_size;
    // ---- input indices (jax pytree, sorted-key flatten of params) ----
    const float* desc0 = (const float*)d_in[0];
    const float* desc1 = (const float*)d_in[1];
    const float* kpts0 = (const float*)d_in[2];
    const float* kpts1 = (const float*)d_in[3];
    const float* sc0   = (const float*)d_in[4];
    const float* sc1   = (const float*)d_in[5];
    const float* alpha = (const float*)d_in[6];            // bin_score
    auto BP = [&](int blk, int off) { return (const float*)d_in[7 + blk * 26 + off]; };
    // per-block leaf offsets (sorted): ca(wk.b,wk.w,wo.b,wo.w,wq.b,wq.w,wv.b,wv.w),
    // ln1(b,g), ln2(b,g), ln3(b,g), mlp(fc1.b,fc1.w,fc2.b,fc2.w), sa(8 like ca)
    const int CA = 0, LN1 = 8, LN2 = 10, LN3 = 12, MLP = 14, SA = 18;
    const float* fproj_b = (const float*)d_in[241];
    const float* fproj_w = (const float*)d_in[242];
    const float* gtok    = (const float*)d_in[243];
    const float* g_wk_b = (const float*)d_in[244];
    const float* g_wk_w = (const float*)d_in[245];
    const float* g_wo_b = (const float*)d_in[246];
    const float* g_wo_w = (const float*)d_in[247];
    const float* g_wq_b = (const float*)d_in[248];
    const float* g_wq_w = (const float*)d_in[249];
    const float* g_wv_b = (const float*)d_in[250];
    const float* g_wv_w = (const float*)d_in[251];
    const float* lnkv_b = (const float*)d_in[252];
    const float* lnkv_g = (const float*)d_in[253];
    const float* lnq_b  = (const float*)d_in[254];
    const float* lnq_g  = (const float*)d_in[255];
    const float* kb0 = (const float*)d_in[256]; const float* kw0 = (const float*)d_in[257];
    const float* kb1 = (const float*)d_in[258]; const float* kw1 = (const float*)d_in[259];
    const float* kb2 = (const float*)d_in[260]; const float* kw2 = (const float*)d_in[261];
    const float* kb3 = (const float*)d_in[262]; const float* kw3 = (const float*)d_in[263];

    // ---- workspace carve ----
    char* wp = (char*)d_ws;
    auto alloc = [&](size_t nbytes) {
        void* r = (void*)wp;
        wp += (nbytes + 255) & ~(size_t)255;
        return r;
    };
    const size_t S_ACT = (size_t)TOTROWS * CC;
    float* xy   = (float*)alloc(S_ACT * 4);
    float* lnb  = (float*)alloc(S_ACT * 4);
    float* qbuf = (float*)alloc(S_ACT * 4);
    float* kbuf = (float*)alloc(S_ACT * 4);
    float* vbuf = (float*)alloc(S_ACT * 4);
    float* aout = (float*)alloc(S_ACT * 4);
    float* hid  = (float*)alloc((size_t)TOTROWS * HID * 4);   // also reused as ln_kv buf
    float* md   = (float*)alloc(S_ACT * 4);
    float* Z    = (float*)alloc((size_t)BB * NP1 * NP1 * 4);
    float* ub   = (float*)alloc((size_t)BB * NP1 * 4);
    float* vb_  = (float*)alloc((size_t)BB * NP1 * 4);
    float* max0 = (float*)alloc((size_t)BB * NPTS * 4);
    float* max1 = (float*)alloc((size_t)BB * NPTS * 4);
    int*   idx0 = (int*)alloc((size_t)BB * NPTS * 4);
    int*   idx1 = (int*)alloc((size_t)BB * NPTS * 4);
    int*   val0 = (int*)alloc((size_t)BB * NPTS * 4);
    float* gatt = (float*)alloc((size_t)VB * NGT * CC * 4);

    float* out = (float*)d_out;
    const size_t OFF_EXPZ = 0;
    const size_t OFF_I0  = (size_t)BB * NP1 * NP1;
    const size_t OFF_I1  = OFF_I0 + BB * NPTS;
    const size_t OFF_MS0 = OFF_I1 + BB * NPTS;
    const size_t OFF_MS1 = OFF_MS0 + BB * NPTS;
    const size_t OFF_G0  = OFF_MS1 + BB * NPTS;
    const size_t OFF_G1  = OFF_G0 + BB * NGT * CC;

    const int R = TOTROWS;
    auto gemm = [&](const float* A, const float* W, const float* bias, const float* res,
                    float* o, int K, int N, int act) {
        dim3 grid((R + 63) / 64, N / 64, 1);
        gemm_kernel<<<grid, 128, 0, stream>>>(A, W, bias, res, o, R, K, N, act);
    };
    auto runLN = [&](const float* in, float* o, const float* g, const float* b) {
        ln_kernel<<<TOTROWS / 8, 256, 0, stream>>>(in, o, g, b, TOTROWS);
    };
    auto runAttn = [&](int kvXor) {
        dim3 grid((RPB + 63) / 64, HEADS, VB);
        attn_kernel<<<grid, 128, 0, stream>>>(qbuf, kbuf, vbuf, aout, kvXor);
    };

    // ---- encoder + token assembly ----
    kenc_kernel<<<(VB * NPTS) / 4, 128, 0, stream>>>(desc0, desc1, kpts0, kpts1, sc0, sc1,
                                                     kb0, kw0, kb1, kw1, kb2, kw2, kb3, kw3, xy);
    set_gt_kernel<<<(VB * NGT * CC + 255) / 256, 256, 0, stream>>>(gtok, xy);

    // ---- transformer blocks ----
    for (int bl = 0; bl < 9; ++bl) {
        // self attention
        runLN(xy, lnb, BP(bl, LN1 + 1), BP(bl, LN1 + 0));
        gemm(lnb, BP(bl, SA + 5), BP(bl, SA + 4), nullptr, qbuf, CC, CC, 0);  // wq
        gemm(lnb, BP(bl, SA + 1), BP(bl, SA + 0), nullptr, kbuf, CC, CC, 0);  // wk
        gemm(lnb, BP(bl, SA + 7), BP(bl, SA + 6), nullptr, vbuf, CC, CC, 0);  // wv
        runAttn(0);
        gemm(aout, BP(bl, SA + 3), BP(bl, SA + 2), xy, xy, CC, CC, 0);        // wo + res
        // cross attention
        runLN(xy, lnb, BP(bl, LN2 + 1), BP(bl, LN2 + 0));
        gemm(lnb, BP(bl, CA + 5), BP(bl, CA + 4), nullptr, qbuf, CC, CC, 0);
        gemm(lnb, BP(bl, CA + 1), BP(bl, CA + 0), nullptr, kbuf, CC, CC, 0);
        gemm(lnb, BP(bl, CA + 7), BP(bl, CA + 6), nullptr, vbuf, CC, CC, 0);
        runAttn(4);
        gemm(aout, BP(bl, CA + 3), BP(bl, CA + 2), xy, xy, CC, CC, 0);
        // mlp
        runLN(xy, lnb, BP(bl, LN3 + 1), BP(bl, LN3 + 0));
        gemm(lnb, BP(bl, MLP + 1), BP(bl, MLP + 0), nullptr, hid, CC, HID, 2); // fc1 + gelu
        gemm(hid, BP(bl, MLP + 3), BP(bl, MLP + 2), xy, xy, HID, CC, 0);       // fc2 + res
    }

    // ---- grouping heads ----
    runLN(xy, lnb, lnq_g, lnq_b);
    gemm(lnb, g_wq_w, g_wq_b, nullptr, qbuf, CC, CC, 0);
    runLN(xy, hid, lnkv_g, lnkv_b);
    gemm(hid, g_wk_w, g_wk_b, nullptr, kbuf, CC, CC, 0);
    gemm(hid, g_wv_w, g_wv_b, nullptr, vbuf, CC, CC, 0);
    {
        dim3 grid(VB, HEADS);
        gattn_kernel<<<grid, 256, 0, stream>>>(qbuf, kbuf, vbuf, gatt);
        dim3 grid2(VB, NGT);
        gout_kernel<<<grid2, CC, 0, stream>>>(gatt, g_wo_w, g_wo_b, xy, out, OFF_G0, OFF_G1);
    }

    // ---- final projection + scores ----
    gemm(xy, fproj_w, fproj_b, nullptr, md, CC, CC, 0);
    {
        dim3 grid(NPTS / 64, NPTS / 64, BB);
        scores_kernel<<<grid, 128, 0, stream>>>(md, Z);
    }
    bins_kernel<<<(BB * 2050 + 255) / 256, 256, 0, stream>>>(Z, alpha);
    uv_init_kernel<<<(BB * NP1 + 255) / 256, 256, 0, stream>>>(ub, vb_);

    // ---- sinkhorn ----
    const float norm = -logf(2048.f);
    const float logn = logf(1024.f) + norm;
    for (int it = 0; it < SINK_IT; ++it) {
        sink_u_kernel<<<BB * NP1, 256, 0, stream>>>(Z, vb_, ub, norm, logn);
        dim3 gv(BB, (NP1 + 127) / 128);
        sink_v_kernel<<<gv, 128, 0, stream>>>(Z, ub, vb_, norm, logn);
    }
    {
        size_t tot = (size_t)BB * NP1 * NP1;
        zfinal_kernel<<<(unsigned)((tot + 255) / 256), 256, 0, stream>>>(Z, ub, vb_,
                                                                         out + OFF_EXPZ, norm);
    }

    // ---- matching ----
    rowmax_kernel<<<BB * NPTS, 256, 0, stream>>>(Z, max0, idx0);
    {
        dim3 gc(BB, (NPTS + 127) / 128);
        colmax_kernel<<<gc, 128, 0, stream>>>(Z, max1, idx1);
    }
    match0_kernel<<<(BB * NPTS + 255) / 256, 256, 0, stream>>>(max0, idx0, idx1,
                                                               out + OFF_MS0, out + OFF_I0, val0);
    match1_kernel<<<(BB * NPTS + 255) / 256, 256, 0, stream>>>(idx0, idx1,
                                                               out + OFF_MS0, val0,
                                                               out + OFF_MS1, out + OFF_I1);
}
